// SophisticatedModel_24575802867742
// MI455X (gfx1250) — compile-verified, matrix operand order checked by reading, not run on hardware
//
#include <hip/hip_runtime.h>
#include <cstddef>

#define HDIM 512
#define LSLOPE 0.2f

typedef __attribute__((ext_vector_type(16))) __bf16 v16bf;
typedef __attribute__((ext_vector_type(8)))  float  v8f;

__device__ __forceinline__ unsigned short f2bf(float f){
  unsigned int u = __float_as_uint(f);
  u = u + 0x7FFFu + ((u >> 16) & 1u);     // round-to-nearest-even
  return (unsigned short)(u >> 16);
}
__device__ __forceinline__ float leakyf(float x){ return x > 0.f ? x : LSLOPE * x; }

// gfx1250 async copy: global -> LDS, 16B per lane, tracked by ASYNCcnt.
__device__ __forceinline__ void async_ld_b128(unsigned lds_byte, const void* gaddr){
  asm volatile("global_load_async_to_lds_b128 %0, %1, off"
               :: "v"(lds_byte), "v"(gaddr) : "memory");
}
__device__ __forceinline__ void wait_async0(){
  asm volatile("s_wait_asynccnt 0" ::: "memory");
}

union FragBF { unsigned int u[8]; v16bf v; };

// ISA 16-bit A layout: lane m=L%16, kh=L/16; dword i holds the K pair
// K = (i>=4?16:0) + kh*8 + (i%4)*2. B mirrored (lane = N%16) from transposed tile.
__device__ __forceinline__ v16bf load_frag(const unsigned short* base, int stride,
                                           int rowBase, int koff, int lane){
  int m = lane & 15, kh = lane >> 4;
  FragBF f;
#pragma unroll
  for (int i = 0; i < 8; ++i){
    int k = ((i & 4) << 2) + kh * 8 + (i & 3) * 2;
    f.u[i] = *(const unsigned int*)(base + (rowBase + m) * stride + koff + k);
  }
  return f.v;
}

// C[M,N] (+)= act( Abf16[M,K] @ B[K,N] + bias ); optional bf16 mirror of C.
// K must be a multiple of 64 (always 512 here). A is bf16 row-major in global,
// staged into LDS via global_load_async_to_lds_b128 (no conversion, ASYNCcnt).
__global__ void __launch_bounds__(256)
gemm_wmma(const unsigned short* __restrict__ Abf, const float* __restrict__ B,
          float* __restrict__ C, unsigned short* __restrict__ Cbf,
          const float* __restrict__ bias,
          int M, int N, int K, int accum, int act)
{
  __shared__ unsigned short As[128 * 64];   // 16 KB, bf16 row-major [m][k]
  __shared__ unsigned short Bst[64 * 64];   // 8 KB,  bf16 transposed [n][k]

  const int m0   = blockIdx.y * 128;
  const int n0   = blockIdx.x * 64;
  const int tid  = threadIdx.x;
  const int lane = tid & 31;
  const int wave = tid >> 5;
  const int wr   = wave >> 1;   // 0..3 -> 32-row strip
  const int wc   = wave & 1;    // 0..1 -> 32-col strip

  const unsigned ldsA = (unsigned)(size_t)(const void*)As;  // low 32 bits = LDS offset

  v8f acc[2][2];
#pragma unroll
  for (int a = 0; a < 2; ++a)
#pragma unroll
    for (int b = 0; b < 2; ++b)
#pragma unroll
      for (int e = 0; e < 8; ++e) acc[a][b][e] = 0.f;

  for (int k0 = 0; k0 < K; k0 += 64){
    // ---- A tile 128x64 bf16 via async copy: 1024 x 16B transfers
#pragma unroll
    for (int i = 0; i < 4; ++i){
      int t   = tid + i * 256;
      int row = t >> 3;               // 8 x 16B segments per 128B row
      int seg = t & 7;
      int grow = m0 + row; if (grow >= M) grow = M - 1;   // clamp: stays in-bounds
      async_ld_b128(ldsA + row * 128 + seg * 16,
                    Abf + (size_t)grow * K + k0 + seg * 8);
    }
    // ---- B tile 64x64 fp32 -> bf16 transposed (weights: tiny, L2 resident)
#pragma unroll
    for (int i = 0; i < 4; ++i){
      int g  = tid + i * 256;         // 1024 groups of 4
      int kr = g >> 4;                // 16 groups per row of 64
      int n4 = (g & 15) << 2;
#pragma unroll
      for (int j = 0; j < 4; ++j){
        int gn = n0 + n4 + j;
        float v = (gn < N) ? B[(size_t)(k0 + kr) * N + gn] : 0.f;
        Bst[(n4 + j) * 64 + kr] = f2bf(v);
      }
    }
    wait_async0();
    __syncthreads();

#pragma unroll
    for (int s = 0; s < 2; ++s){
      v16bf a0 = load_frag(As,  64, wr * 32,      s * 32, lane);
      v16bf a1 = load_frag(As,  64, wr * 32 + 16, s * 32, lane);
      v16bf b0 = load_frag(Bst, 64, wc * 32,      s * 32, lane);
      v16bf b1 = load_frag(Bst, 64, wc * 32 + 16, s * 32, lane);
      acc[0][0] = __builtin_amdgcn_wmma_f32_16x16x32_bf16(false, a0, false, b0, (short)0, acc[0][0], false, false);
      acc[0][1] = __builtin_amdgcn_wmma_f32_16x16x32_bf16(false, a0, false, b1, (short)0, acc[0][1], false, false);
      acc[1][0] = __builtin_amdgcn_wmma_f32_16x16x32_bf16(false, a1, false, b0, (short)0, acc[1][0], false, false);
      acc[1][1] = __builtin_amdgcn_wmma_f32_16x16x32_bf16(false, a1, false, b1, (short)0, acc[1][1], false, false);
    }
    __syncthreads();
  }

  // ---- store: C/D layout -> lane n=L%16, dword e maps to row e + 8*(L/16)
  const int nn   = lane & 15;
  const int half = lane >> 4;
#pragma unroll
  for (int sm = 0; sm < 2; ++sm)
#pragma unroll
    for (int sn = 0; sn < 2; ++sn)
#pragma unroll
      for (int e = 0; e < 8; ++e){
        int row = m0 + wr * 32 + sm * 16 + half * 8 + e;
        int col = n0 + wc * 32 + sn * 16 + nn;
        if (row < M && col < N){
          float val = acc[sm][sn][e];
          if (bias)  val += bias[col];
          size_t idx = (size_t)row * N + col;
          if (accum) val += C[idx];
          if (act)   val = leakyf(val);
          C[idx] = val;
          if (Cbf) Cbf[idx] = f2bf(val);
        }
      }
}

// -------- softmax pooling helpers --------
__global__ void k_logits(const float* __restrict__ xk, const float* __restrict__ seed,
                         float* __restrict__ out, int n){
  int row  = blockIdx.x * 8 + (threadIdx.x >> 5);
  int lane = threadIdx.x & 31;
  if (row >= n) return;
  float s = 0.f;
  for (int k = lane; k < HDIM; k += 32) s += xk[(size_t)row * HDIM + k] * seed[k];
#pragma unroll
  for (int off = 16; off > 0; off >>= 1) s += __shfl_down(s, off, 32);
  if (lane == 0) out[row] = s * 0.04419417382415922f;   // 1/sqrt(512)
}

__global__ void k_rmax1(const float* __restrict__ v, float* __restrict__ part, int n){
  __shared__ float sm[256];
  float m = -3.4e38f;
  for (int i = blockIdx.x * 256 + threadIdx.x; i < n; i += gridDim.x * 256) m = fmaxf(m, v[i]);
  sm[threadIdx.x] = m; __syncthreads();
  for (int s = 128; s > 0; s >>= 1){ if (threadIdx.x < s) sm[threadIdx.x] = fmaxf(sm[threadIdx.x], sm[threadIdx.x + s]); __syncthreads(); }
  if (threadIdx.x == 0) part[blockIdx.x] = sm[0];
}
__global__ void k_rmax2(const float* __restrict__ part, float* __restrict__ out, int n){
  __shared__ float sm[256];
  float m = -3.4e38f;
  for (int i = threadIdx.x; i < n; i += 256) m = fmaxf(m, part[i]);
  sm[threadIdx.x] = m; __syncthreads();
  for (int s = 128; s > 0; s >>= 1){ if (threadIdx.x < s) sm[threadIdx.x] = fmaxf(sm[threadIdx.x], sm[threadIdx.x + s]); __syncthreads(); }
  if (threadIdx.x == 0) out[0] = sm[0];
}
__global__ void k_expsum1(float* __restrict__ v, const float* __restrict__ mx,
                          float* __restrict__ part, int n){
  __shared__ float sm[256];
  float m = mx[0], s = 0.f;
  for (int i = blockIdx.x * 256 + threadIdx.x; i < n; i += gridDim.x * 256){
    float e = __expf(v[i] - m); v[i] = e; s += e;
  }
  sm[threadIdx.x] = s; __syncthreads();
  for (int st = 128; st > 0; st >>= 1){ if (threadIdx.x < st) sm[threadIdx.x] += sm[threadIdx.x + st]; __syncthreads(); }
  if (threadIdx.x == 0) part[blockIdx.x] = sm[0];
}
__global__ void k_rsum2(const float* __restrict__ part, float* __restrict__ out, int n){
  __shared__ float sm[256];
  float s = 0.f;
  for (int i = threadIdx.x; i < n; i += 256) s += part[i];
  sm[threadIdx.x] = s; __syncthreads();
  for (int st = 128; st > 0; st >>= 1){ if (threadIdx.x < st) sm[threadIdx.x] += sm[threadIdx.x + st]; __syncthreads(); }
  if (threadIdx.x == 0) out[0] = sm[0];
}
// g_raw[h] += sum_i e[i]*xv[i,h]   (unnormalized; /sumExp folded into k_gterm)
__global__ void k_gv(const float* __restrict__ xv, const float* __restrict__ e,
                     float* __restrict__ graw, int n){
  float a0 = 0.f, a1 = 0.f;
  int c0 = threadIdx.x, c1 = threadIdx.x + 256;
  for (int i = blockIdx.x; i < n; i += gridDim.x){
    float w = e[i];
    a0 += w * xv[(size_t)i * HDIM + c0];
    a1 += w * xv[(size_t)i * HDIM + c1];
  }
  atomicAdd(&graw[c0], a0);
  atomicAdd(&graw[c1], a1);
}
// gterm[j] = (sum_k g_raw[k]*Wg[k,j]) / sumExp
__global__ void k_gterm(const float* __restrict__ graw, const float* __restrict__ Wg,
                        const float* __restrict__ sum, float* __restrict__ gt){
  float inv = 1.0f / sum[0];
  for (int j = threadIdx.x; j < HDIM; j += 256){
    float s = 0.f;
    for (int k = 0; k < HDIM; ++k) s += graw[k] * Wg[k * HDIM + j];
    gt[j] = s * inv;
  }
}

// -------- edge scatter-mean --------
__global__ void k_scatter(const int* __restrict__ src, const int* __restrict__ dst,
                          const float* __restrict__ x, const float* __restrict__ pos,
                          const float* __restrict__ Wpos, float* __restrict__ agg,
                          float* __restrict__ cnt, int E){
  int e = blockIdx.x * 8 + (threadIdx.x >> 5);
  if (e >= E) return;
  int lane = threadIdx.x & 31;
  int s = src[e], d = dst[e];
  float dx = pos[(size_t)s * 2]     - pos[(size_t)d * 2];
  float dy = pos[(size_t)s * 2 + 1] - pos[(size_t)d * 2 + 1];
  for (int h = lane; h < HDIM; h += 32){
    float m = x[(size_t)s * HDIM + h] + dx * Wpos[h] + dy * Wpos[HDIM + h];
    atomicAdd(&agg[(size_t)d * HDIM + h], m);
  }
  if (lane == 0) atomicAdd(&cnt[d], 1.0f);
}
// scatter-mean finalize, emitted directly as bf16 (GEMM A operand)
__global__ void k_div_bf(const float* __restrict__ agg, const float* __restrict__ cnt,
                         unsigned short* __restrict__ o, int total){
  for (int i = blockIdx.x * 256 + threadIdx.x; i < total; i += gridDim.x * 256)
    o[i] = f2bf(agg[i] / fmaxf(cnt[i >> 9], 1.0f));
}
// x = leaky(acc/NREL + gterm), fp32 (for edge gather) + bf16 mirror (for GEMMs)
__global__ void k_fin(const float* __restrict__ acc, const float* __restrict__ gt,
                      float* __restrict__ x, unsigned short* __restrict__ xbf, int total){
  for (int i = blockIdx.x * 256 + threadIdx.x; i < total; i += gridDim.x * 256){
    float v = leakyf(acc[i] * (1.0f / 3.0f) + gt[i & (HDIM - 1)]);
    x[i] = v;
    xbf[i] = f2bf(v);
  }
}
__global__ void k_cvt(const float* __restrict__ a, unsigned short* __restrict__ o, int n){
  for (int i = blockIdx.x * 256 + threadIdx.x; i < n; i += gridDim.x * 256) o[i] = f2bf(a[i]);
}
__global__ void k_add3(const float* __restrict__ a, const float* __restrict__ b,
                       const float* __restrict__ c, float* __restrict__ o, int n){
  int i = blockIdx.x * 256 + threadIdx.x;
  if (i < n) o[i] = a[i] + b[i] + c[i];
}

extern "C" void kernel_launch(void* const* d_in, const int* in_sizes, int n_in,
                              void* d_out, int out_size, void* d_ws, size_t ws_size,
                              hipStream_t stream) {
  const float* x_window = (const float*)d_in[0];
  const float* x_pos    = (const float*)d_in[1];
  const int*   e_near   = (const int*)d_in[2];
  const int*   e_close  = (const int*)d_in[3];
  const int*   e_sim    = (const int*)d_in[4];
  const float* W_pre    = (const float*)d_in[5];
  const float* W_post   = (const float*)d_in[6];
  const float* Wl       = (const float*)d_in[7];
  const float* Wr       = (const float*)d_in[8];
  const float* Wpos     = (const float*)d_in[9];
  const float* Wg       = (const float*)d_in[10];
  const float* seed     = (const float*)d_in[11];
  const float* Wk       = (const float*)d_in[12];
  const float* Wv       = (const float*)d_in[13];
  const float* W_out    = (const float*)d_in[14];
  const float* b_out    = (const float*)d_in[15];

  const int N    = in_sizes[0] / HDIM;
  const int E    = in_sizes[2] / 2;
  const int OUTD = in_sizes[15];
  const int L = 4, NREL = 3;
  const size_t NH = (size_t)N * HDIM;
  const int HH = HDIM * HDIM;

  float* ws0  = (float*)d_ws;
  float* xbuf = ws0;                  // N*H fp32 current features (edge gather)
  float* t0   = ws0 + NH;             // N*H fp32 scratch (xk / xv / agg accum)
  float* accb = ws0 + 2 * NH;         // N*H fp32 layer accumulator
  unsigned short* xbf   = (unsigned short*)(ws0 + 3 * NH);  // N*H bf16 mirror of x
  unsigned short* aggbf = xbf + NH;                         // N*H bf16 agg / temp
  float* fl   = (float*)(aggbf + NH);
  float* lbuf = fl;                   // N   logits -> exp
  float* cnt  = lbuf + N;             // N   edge counts
  float* part = cnt + N;              // 256 reduce partials
  float* scal = part + 256;           // 2   {max, sumExp}
  float* graw = scal + 2;             // H
  float* gt   = graw + HDIM;          // H
  float* wrs  = gt + HDIM;            // H*H Wr[l,0]+Wr[l,1]+Wr[l,2]

  auto gemm = [&](const unsigned short* A, const float* B, float* C,
                  unsigned short* Cbf, const float* bias,
                  int M, int Nc, int K, int accum, int act){
    dim3 grid((Nc + 63) / 64, (M + 127) / 128);
    gemm_wmma<<<grid, 256, 0, stream>>>(A, B, C, Cbf, bias, M, Nc, K, accum, act);
  };

  // bootstrap: bf16 mirror of x_window
  k_cvt<<<4096, 256, 0, stream>>>(x_window, xbf, (int)NH);

  // x = leaky(leaky(xw @ W_pre) @ W_post); aggbf reused as bf16 temp of the mid value
  gemm(xbf,   W_pre,  t0,   aggbf, nullptr, N, HDIM, HDIM, 0, 1);
  gemm(aggbf, W_post, xbuf, xbf,   nullptr, N, HDIM, HDIM, 0, 1);

  for (int l = 0; l < L; ++l){
    // GMPool_I: softmax((x@Wk)·seed / sqrt(H)), g = attn @ (x@Wv), gterm = g@Wg
    gemm(xbf, Wk + (size_t)l * HH, t0, nullptr, nullptr, N, HDIM, HDIM, 0, 0);
    k_logits <<<(N + 7) / 8, 256, 0, stream>>>(t0, seed + (size_t)l * HDIM, lbuf, N);
    k_rmax1  <<<256, 256, 0, stream>>>(lbuf, part, N);
    k_rmax2  <<<1,   256, 0, stream>>>(part, scal, 256);
    k_expsum1<<<256, 256, 0, stream>>>(lbuf, scal, part, N);
    k_rsum2  <<<1,   256, 0, stream>>>(part, scal + 1, 256);
    gemm(xbf, Wv + (size_t)l * HH, t0, nullptr, nullptr, N, HDIM, HDIM, 0, 0);
    hipMemsetAsync(graw, 0, HDIM * sizeof(float), stream);
    k_gv   <<<512, 256, 0, stream>>>(t0, lbuf, graw, N);
    k_gterm<<<1,   256, 0, stream>>>(graw, Wg + (size_t)l * HH, scal + 1, gt);

    // acc = x @ (Wr[l,0]+Wr[l,1]+Wr[l,2])   (self-transform folded to one GEMM)
    k_add3<<<(HH + 255) / 256, 256, 0, stream>>>(
        Wr + ((size_t)l * NREL + 0) * HH,
        Wr + ((size_t)l * NREL + 1) * HH,
        Wr + ((size_t)l * NREL + 2) * HH, wrs, HH);
    gemm(xbf, wrs, accb, nullptr, nullptr, N, HDIM, HDIM, 0, 0);

    // per relation: scatter-mean (atomics) -> bf16 agg -> acc += agg @ Wl[l,r]
    for (int r = 0; r < NREL; ++r){
      const int* eg = (r == 0) ? e_near : (r == 1) ? e_close : e_sim;
      hipMemsetAsync(t0,  0, NH * sizeof(float), stream);
      hipMemsetAsync(cnt, 0, (size_t)N * sizeof(float), stream);
      k_scatter<<<(E + 7) / 8, 256, 0, stream>>>(
          eg, eg + E, xbuf, x_pos,
          Wpos + ((size_t)l * NREL + r) * 2 * HDIM, t0, cnt, E);
      k_div_bf<<<4096, 256, 0, stream>>>(t0, cnt, aggbf, (int)NH);
      gemm(aggbf, Wl + ((size_t)l * NREL + r) * HH, accb, nullptr, nullptr,
           N, HDIM, HDIM, 1, 0);
    }

    // x = leaky(acc/NREL + gterm)   (gterm appears NREL times in acc sum)
    k_fin<<<4096, 256, 0, stream>>>(accb, gt, xbuf, xbf, (int)NH);
  }

  // out = x @ W_out + b_out
  gemm(xbf, W_out, (float*)d_out, nullptr, b_out, N, OUTD, HDIM, 0, 0);
}
